// CausalSelfAttention_88399016886820
// MI455X (gfx1250) — matrix operand order repaired and checked by reading, not compiled, not run on hardware
//
#include <hip/hip_runtime.h>
#include <hip/hip_bf16.h>

// ---------------------------------------------------------------------------
// CDNA5 / gfx1250 fused attention layer.
// Memory-bound workload (KV concat + KV streaming ~1.6 GB total traffic at
// 23.3 TB/s); matmuls done with v_wmma_f32_16x16x32_f16 (f16 in, f32 acc).
// ---------------------------------------------------------------------------

typedef _Float16 half_t;
typedef __attribute__((ext_vector_type(16))) _Float16 v16h;
typedef __attribute__((ext_vector_type(8)))  float    v8f;

#define D_MODEL  1024
#define N_HEADS  16
#define HEAD_DIM 64
#define BATCH    16
#define SEQ      16
#define PAST     4096
#define TOTKV    4112                       // PAST + SEQ
#define MROWS    256                        // BATCH*SEQ
#define Y_ELEMS  ((size_t)MROWS * D_MODEL)  // 262144
#define KV_PER_BH ((size_t)TOTKV * HEAD_DIM)        // 263168 floats per (b,h)
#define KV_ELEMS  ((size_t)256 * KV_PER_BH)         // 67371008
#define KBASE     (Y_ELEMS)
#define VBASE     (Y_ELEMS + KV_ELEMS)

// ------------------------------ WMMA helpers -------------------------------

__device__ __forceinline__ v8f wmma16(v16h a, v16h b, v8f c) {
  // 8 args: (neg_a, A, neg_b, B, c_mod, C, reuse_a, reuse_b)
  return __builtin_amdgcn_wmma_f32_16x16x32_f16(false, a, false, b,
                                                (short)0, c, false, false);
}

__device__ __forceinline__ v8f zero8() {
  v8f z = {0.f, 0.f, 0.f, 0.f, 0.f, 0.f, 0.f, 0.f};
  return z;
}

// A-fragment (16x32 f16). Layout: lanes 0-15 M=0..15 hold K=0..15,
// lanes 16-31 (same M=lane%16) hold K=16..31; element e of v16h is
// VGPR e/2, half e%2.
__device__ __forceinline__ v16h load_A_f32m(const float* base, int ld,
                                            int m0, int k0, int lane) {
  const int row = lane & 15;
  const int kb  = (lane < 16) ? 0 : 16;
  const float* p = base + (size_t)(m0 + row) * ld + (k0 + kb);
  v16h a;
#pragma unroll
  for (int e = 0; e < 8; ++e) a[e] = (half_t)p[e];
#pragma unroll
  for (int e = 0; e < 8; ++e) a[8 + e] = (half_t)p[8 + e];
  return a;
}

__device__ __forceinline__ v16h load_A_h(const half_t* base, int ld,
                                         int m0, int k0, int lane) {
  const int row = lane & 15;
  const int kb  = (lane < 16) ? 0 : 16;
  const half_t* p = base + (size_t)(m0 + row) * ld + (k0 + kb);
  v16h a;
#pragma unroll
  for (int e = 0; e < 8; ++e) a[e] = p[e];
#pragma unroll
  for (int e = 0; e < 8; ++e) a[8 + e] = p[8 + e];
  return a;
}

// B-fragment (32x16 f16), B[k][n] = src[n0+n][k0+k] (i.e. the "W^T" / K^T use
// case: row-major rows supply the contraction dim). lane -> N=lane%16,
// K split: lanes 0-15 K=0..15, lanes 16-31 K=16..31.
__device__ __forceinline__ v16h load_B_rm(const float* w, int ld, int n0,
                                          int k0, int lane, int rowmax) {
  int row = n0 + (lane & 15);
  row = min(row, rowmax);
  const int kb = (lane < 16) ? 0 : 16;
  const float* p = w + (size_t)row * ld + (k0 + kb);
  v16h b;
#pragma unroll
  for (int e = 0; e < 16; ++e) b[e] = (half_t)p[e];
  return b;
}

// B-fragment where B[k][n] = src[row0+k][col0+n] (the V case).
__device__ __forceinline__ v16h load_B_cm(const float* m, int ld, int row0,
                                          int col0, int lane, int rowmax) {
  const int col = col0 + (lane & 15);
  const int kb  = (lane < 16) ? 0 : 16;
  v16h b;
#pragma unroll
  for (int e = 0; e < 16; ++e) {
    int rr = min(row0 + kb + e, rowmax);
    b[e] = (half_t)m[(size_t)rr * ld + col];
  }
  return b;
}

// half-wave (16 lane) reductions: xor masks < 16 never cross half-waves.
__device__ __forceinline__ float half16_max(float v) {
#pragma unroll
  for (int m = 1; m < 16; m <<= 1) v = fmaxf(v, __shfl_xor(v, m, 32));
  return v;
}
__device__ __forceinline__ float half16_sum(float v) {
#pragma unroll
  for (int m = 1; m < 16; m <<= 1) v += __shfl_xor(v, m, 32);
  return v;
}

// --------------------------- 1) KV cache concat ----------------------------
// 512 blocks; block = one (region,b,h) 1MB contiguous slab. Async LDS-staged
// DMA path when the gfx1250 async builtins exist, float4 fallback otherwise.

#if __has_builtin(__builtin_amdgcn_global_load_async_to_lds_b128) && \
    __has_builtin(__builtin_amdgcn_global_store_async_from_lds_b128)
#define USE_ASYNC_LDS 1
// Exact param types per hipcc diagnostics:
//   arg0: int __vector_size__(16) __device__*  (addrspace(1))
//   arg1: int __vector_size__(16) __shared__*  (addrspace(3))
typedef int v4i_vs __attribute__((vector_size(16)));
typedef __attribute__((address_space(1))) v4i_vs* as1v;
typedef __attribute__((address_space(3))) v4i_vs* as3v;
#endif

__global__ void __launch_bounds__(256) copy_kv_kernel(const float* __restrict__ ck,
                                                      const float* __restrict__ cv,
                                                      float* __restrict__ out) {
  const int blkid  = blockIdx.x;        // 0..511
  const int region = blkid >> 8;        // 0 = K, 1 = V
  const int bh     = blkid & 255;
  const float* src = (region ? cv : ck) + (size_t)bh * (PAST * HEAD_DIM);
  float* dst = out + (region ? VBASE : KBASE) + (size_t)bh * KV_PER_BH;

#ifdef USE_ASYNC_LDS
  __shared__ char stage[8][8192];
  const int wave = threadIdx.x >> 5;
  const int lane = threadIdx.x & 31;
  // 1 MB / 8 waves = 128 KB per wave, moved in 16 chunks of 8 KB through LDS.
  const char* gs = (const char*)src + (size_t)wave * 131072;
  char*       gd = (char*)dst + (size_t)wave * 131072;
  char*       lb = &stage[wave][0];
#pragma unroll 1
  for (int c = 0; c < 16; ++c) {
#pragma unroll
    for (int i = 0; i < 16; ++i) {
      __builtin_amdgcn_global_load_async_to_lds_b128(
          (as1v)(void*)(gs + (size_t)c * 8192 + i * 512 + lane * 16),
          (as3v)(void*)(lb + i * 512 + lane * 16), 0, 0);
    }
#if __has_builtin(__builtin_amdgcn_s_wait_asynccnt)
    __builtin_amdgcn_s_wait_asynccnt(0);
#else
    asm volatile("s_wait_asynccnt 0x0" ::: "memory");
#endif
#pragma unroll
    for (int i = 0; i < 16; ++i) {
      __builtin_amdgcn_global_store_async_from_lds_b128(
          (as1v)(void*)(gd + (size_t)c * 8192 + i * 512 + lane * 16),
          (as3v)(void*)(lb + i * 512 + lane * 16), 0, 0);
    }
#if __has_builtin(__builtin_amdgcn_s_wait_asynccnt)
    __builtin_amdgcn_s_wait_asynccnt(0);
#else
    asm volatile("s_wait_asynccnt 0x0" ::: "memory");
#endif
  }
#else
  const float4* s4 = (const float4*)src;
  float4*       d4 = (float4*)dst;
  for (int i = threadIdx.x; i < (PAST * HEAD_DIM) / 4; i += 256) d4[i] = s4[i];
#endif
}

// --------------------------- 2) QKV projection -----------------------------
// y = x @ W^T + b, M=256, N=3072, K=1024. Grid: 192 n-tiles x 4 waves
// (wave = 64-row m-block, reuses each B fragment 4x). q -> ws (f16, *0.125),
// k/v -> d_out tail rows [4096..4111] as f32.

__global__ void __launch_bounds__(128) qkv_kernel(const float* __restrict__ x,
                                                  const float* __restrict__ w,
                                                  const float* __restrict__ bias,
                                                  float* __restrict__ out,
                                                  half_t* __restrict__ q_ws) {
  const int lane = threadIdx.x & 31;
  const int wave = threadIdx.x >> 5;
  const int n0   = blockIdx.x * 16;
  const int m0   = wave * 64;

  v8f c[4];
#pragma unroll
  for (int t = 0; t < 4; ++t) c[t] = zero8();

  for (int k0 = 0; k0 < D_MODEL; k0 += 32) {
    v16h bf = load_B_rm(w, D_MODEL, n0, k0, lane, 3 * D_MODEL - 1);
#pragma unroll
    for (int t = 0; t < 4; ++t) {
      v16h af = load_A_f32m(x, D_MODEL, m0 + t * 16, k0, lane);
      c[t] = wmma16(af, bf, c[t]);
    }
  }

  const int n  = lane & 15;
  const float bv = bias[n0 + n];
  const int cls = n0 >> 10;            // 0=q 1=k 2=v
  const int h   = (n0 & 1023) >> 6;
  const int d   = (n0 & 63) + n;
  const int mh  = (lane < 16) ? 0 : 8;
#pragma unroll
  for (int t = 0; t < 4; ++t) {
#pragma unroll
    for (int r = 0; r < 8; ++r) {
      const int m  = m0 + t * 16 + r + mh;
      const int bb = m >> 4, s = m & 15;
      const float val = c[t][r] + bv;
      if (cls == 0) {
        q_ws[(((size_t)(bb * 16 + h) * 16 + s) << 6) + d] = (half_t)(val * 0.125f);
      } else {
        const size_t base = (cls == 1) ? KBASE : VBASE;
        out[base + (((size_t)(bb * 16 + h) * TOTKV + PAST + s) << 6) + d] = val;
      }
    }
  }
}

// ----------------------------- 3) attention --------------------------------
// One workgroup (4 waves) per (b,h). 129 key-blocks of 32 split across waves,
// per-wave flash-softmax state, P transposed through per-wave LDS, partials
// merged in LDS at the end.

__global__ void __launch_bounds__(128) attn_kernel(const float* __restrict__ out,
                                                   const half_t* __restrict__ q_ws,
                                                   half_t* __restrict__ o_ws) {
  __shared__ half_t p_lds[4][16 * 32];
  __shared__ float  red_max[4][16];
  __shared__ float  red_sum[4][16];
  __shared__ float  red_o[4][16][64];

  const int lane = threadIdx.x & 31;
  const int wave = threadIdx.x >> 5;
  const int bh = blockIdx.x, bb = bh >> 4, h = bh & 15;
  const float*  Kp = out + KBASE + (size_t)bh * KV_PER_BH;
  const float*  Vp = out + VBASE + (size_t)bh * KV_PER_BH;
  const half_t* Qp = q_ws + (size_t)bh * (SEQ * HEAD_DIM);

  const v16h aq0 = load_A_h(Qp, HEAD_DIM, 0, 0, lane);
  const v16h aq1 = load_A_h(Qp, HEAD_DIM, 0, 32, lane);

  v8f o[4];
#pragma unroll
  for (int t = 0; t < 4; ++t) o[t] = zero8();
  float rmax[8], rsum[8];
#pragma unroll
  for (int r = 0; r < 8; ++r) { rmax[r] = -3.0e38f; rsum[r] = 0.f; }

  const int mh = (lane < 16) ? 0 : 8;
  const int n  = lane & 15;
  half_t* pl = &p_lds[wave][0];

  for (int blk = wave; blk < 129; blk += 4) {
    const int key0 = blk * 32;
    if (key0 + 128 < TOTKV) {               // prefetch this wave's next block
      __builtin_prefetch(Kp + (size_t)(key0 + 128) * HEAD_DIM + lane * 8, 0, 1);
      __builtin_prefetch(Vp + (size_t)(key0 + 128) * HEAD_DIM + lane * 8, 0, 1);
    }
    // S = (Q/8) K^T : two 16x16 tiles over 32 keys
    v8f s[2];
#pragma unroll
    for (int t = 0; t < 2; ++t) {
      v16h bk0 = load_B_rm(Kp, HEAD_DIM, key0 + t * 16, 0, lane, TOTKV - 1);
      v16h bk1 = load_B_rm(Kp, HEAD_DIM, key0 + t * 16, 32, lane, TOTKV - 1);
      v8f acc = zero8();
      acc = wmma16(aq0, bk0, acc);
      acc = wmma16(aq1, bk1, acc);
      if (key0 + t * 16 + n >= TOTKV) {     // mask padded tail keys
#pragma unroll
        for (int r = 0; r < 8; ++r) acc[r] = -3.0e38f;
      }
      s[t] = acc;
    }
    // running softmax per row (row r+mh lives in reg r, its half-wave)
    float pr0[8], pr1[8];
#pragma unroll
    for (int r = 0; r < 8; ++r) {
      const float mx   = half16_max(fmaxf(s[0][r], s[1][r]));
      const float mnew = fmaxf(rmax[r], mx);
      const float scl  = __expf(rmax[r] - mnew);
      const float p0 = __expf(s[0][r] - mnew);
      const float p1 = __expf(s[1][r] - mnew);
      rsum[r] = rsum[r] * scl + half16_sum(p0 + p1);
      rmax[r] = mnew;
#pragma unroll
      for (int dt = 0; dt < 4; ++dt) o[dt][r] *= scl;
      pr0[r] = p0; pr1[r] = p1;
    }
    // C-layout -> A-layout transpose of P through per-wave LDS
#pragma unroll
    for (int r = 0; r < 8; ++r) {
      pl[(r + mh) * 32 + n]      = (half_t)pr0[r];
      pl[(r + mh) * 32 + 16 + n] = (half_t)pr1[r];
    }
    asm volatile("" ::: "memory");          // keep DS store->load order
    const v16h pa = load_A_h(pl, 32, 0, 0, lane);
    // O += P V
#pragma unroll
    for (int dt = 0; dt < 4; ++dt) {
      v16h bv = load_B_cm(Vp, HEAD_DIM, key0, dt * 16, lane, TOTKV - 1);
      o[dt] = wmma16(pa, bv, o[dt]);
    }
  }

  // publish per-wave partials
#pragma unroll
  for (int r = 0; r < 8; ++r) {
    const int m = r + mh;
    if (n == 0) { red_max[wave][m] = rmax[r]; red_sum[wave][m] = rsum[r]; }
#pragma unroll
    for (int dt = 0; dt < 4; ++dt) red_o[wave][m][dt * 16 + n] = o[dt][r];
  }
  __syncthreads();

  // merge 4 partials; write o (f16) in [b,s,h*64+d] layout for the out-proj
  for (int idx = threadIdx.x; idx < SEQ * HEAD_DIM; idx += 128) {
    const int m = idx >> 6, d = idx & 63;
    float gm = red_max[0][m];
#pragma unroll
    for (int wv = 1; wv < 4; ++wv) gm = fmaxf(gm, red_max[wv][m]);
    float tot = 0.f, val = 0.f;
#pragma unroll
    for (int wv = 0; wv < 4; ++wv) {
      const float e = __expf(red_max[wv][m] - gm);
      tot += red_sum[wv][m] * e;
      val += red_o[wv][m][d] * e;
    }
    o_ws[(size_t)(bb * 16 + m) * D_MODEL + h * 64 + d] = (half_t)(val / tot);
  }
}

// --------------------------- 4) output projection --------------------------

__global__ void __launch_bounds__(128) oproj_kernel(const half_t* __restrict__ o_ws,
                                                    const float* __restrict__ w,
                                                    const float* __restrict__ bias,
                                                    float* __restrict__ y) {
  const int lane = threadIdx.x & 31;
  const int wave = threadIdx.x >> 5;
  const int n0   = blockIdx.x * 16;
  const int m0   = wave * 64;

  v8f c[4];
#pragma unroll
  for (int t = 0; t < 4; ++t) c[t] = zero8();

  for (int k0 = 0; k0 < D_MODEL; k0 += 32) {
    v16h bf = load_B_rm(w, D_MODEL, n0, k0, lane, D_MODEL - 1);
#pragma unroll
    for (int t = 0; t < 4; ++t) {
      v16h af = load_A_h(o_ws, D_MODEL, m0 + t * 16, k0, lane);
      c[t] = wmma16(af, bf, c[t]);
    }
  }

  const int n = lane & 15;
  const float bv = bias[n0 + n];
  const int mh = (lane < 16) ? 0 : 8;
#pragma unroll
  for (int t = 0; t < 4; ++t) {
#pragma unroll
    for (int r = 0; r < 8; ++r) {
      const int m = m0 + t * 16 + r + mh;
      y[(size_t)m * D_MODEL + n0 + n] = c[t][r] + bv;
    }
  }
}

// ------------------------------- launcher ----------------------------------

extern "C" void kernel_launch(void* const* d_in, const int* in_sizes, int n_in,
                              void* d_out, int out_size, void* d_ws, size_t ws_size,
                              hipStream_t stream) {
  (void)in_sizes; (void)n_in; (void)out_size; (void)ws_size;
  const float* x  = (const float*)d_in[0];
  const float* ck = (const float*)d_in[1];
  const float* cv = (const float*)d_in[2];
  const float* qw = (const float*)d_in[3];
  const float* qb = (const float*)d_in[4];
  const float* ow = (const float*)d_in[5];
  const float* ob = (const float*)d_in[6];
  float* out = (float*)d_out;
  half_t* q_ws = (half_t*)d_ws;                       // 262144 halves (512 KB)
  half_t* o_ws = q_ws + Y_ELEMS;                      // 262144 halves (512 KB)

  copy_kv_kernel<<<512, 256, 0, stream>>>(ck, cv, out);
  qkv_kernel<<<3 * D_MODEL / 16, 128, 0, stream>>>(x, qw, qb, out, q_ws);
  attn_kernel<<<BATCH * N_HEADS, 128, 0, stream>>>(out, q_ws, o_ws);
  oproj_kernel<<<D_MODEL / 16, 128, 0, stream>>>(o_ws, ow, ob, out);
}
